// TTEqNetForImages_8126078124308
// MI455X (gfx1250) — matrix operand order repaired and checked by reading, not compiled, not run on hardware
//
#include <hip/hip_runtime.h>
#include <hip/hip_bf16.h>

typedef __attribute__((ext_vector_type(16))) __bf16 v16bf;
typedef __attribute__((ext_vector_type(8)))  float  v8f;

namespace {
constexpr int    kB     = 16384;
constexpr int    kF     = 64;
constexpr int    kH     = 64;
constexpr int    kD     = 64;
constexpr int    kC     = 1000;
constexpr int    kCp    = 1024;          // padded class count
constexpr int    kSteps = 62;
constexpr size_t kRF    = (size_t)kB * kF;   // 2^20 (b,f) rows

// ---- workspace layout (bytes); total ~150 MB ----
constexpr size_t XT_OFF = 0;                         // bf16 xt[d][b*64+f]
constexpr size_t XT_SZ  = (size_t)kD * kRF * 2;      // 128 MB
constexpr size_t BM_OFF = XT_OFF + XT_SZ;            // bf16, fragment-packed
constexpr size_t BM_SZ  = (size_t)kH * kF * kH * 2;  // 512 KB
constexpr size_t BL_OFF = BM_OFF + BM_SZ;            // bf16, fragment-packed
constexpr size_t BL_SZ  = (size_t)kH * kF * kCp * 2; // 8 MB
constexpr size_t BF_OFF = BL_OFF + BL_SZ;            // bf16, fragment-packed
constexpr size_t BF_SZ  = (size_t)kF * kH * 2;       // 8 KB
constexpr size_t Z_OFF  = BF_OFF + BF_SZ;            // f32 z[b][h]
constexpr size_t Z_SZ   = (size_t)kB * kH * 4;       // 4 MB
constexpr size_t CR_OFF = Z_OFF + Z_SZ;              // bf16 carry[b][h]
constexpr size_t CR_SZ  = (size_t)kB * kH * 2;       // 2 MB
constexpr size_t ST_OFF = CR_OFF + CR_SZ;            // f32 stats[62][128]
} // namespace

// ---------------- WMMA helpers --------------------------------------------

__device__ __forceinline__ v8f wmma_bf16(v16bf a, v16bf b, v8f c) {
  return __builtin_amdgcn_wmma_f32_16x16x32_bf16(
      /*neg_a=*/false, a, /*neg_b=*/false, b,
      /*c_mod=*/(short)0, c, /*reuse_a=*/false, /*reuse_b=*/false);
}

// A 16x32 bf16 fragment = one contiguous 32B run per lane:
// lanes 0-15 (row M=lane): K {0..15};  lanes 16-31: K {16..31}.
__device__ __forceinline__ void load_a_frags(const __bf16* __restrict__ frow,
                                             int lane, v16bf& a0, v16bf& a1) {
  const uint4* p = (const uint4*)frow + ((lane & 16) ? 2 : 0);
  union { uint4 u[2]; v16bf v; } U0, U1;
  U0.u[0] = p[0]; U0.u[1] = p[1];   // chunk 0: f 0..31
  U1.u[0] = p[4]; U1.u[1] = p[5];   // chunk 1: f 32..63
  a0 = U0.v; a1 = U1.v;
}

// B fragment: 32 contiguous bytes in LDS (fragment-packed layout).
__device__ __forceinline__ v16bf load_frag_lds(const __bf16* __restrict__ p) {
  union { uint4 u[2]; v16bf v; } U;
  U.u[0] = ((const uint4*)p)[0];
  U.u[1] = ((const uint4*)p)[1];
  return U.v;
}

// Splat one bf16 across a v16bf as 8 identical dwords so the packed bf16
// multiply can reuse a single VGPR (avoids 16x v_mov_b16).
__device__ __forceinline__ v16bf splat_bf16_bits(unsigned short s) {
  const unsigned pp = (unsigned)s * 0x00010001u;
  union { unsigned u[8]; v16bf v; } U;
#pragma unroll
  for (int i = 0; i < 8; ++i) U.u[i] = pp;
  return U.v;
}

// ---------------- prep kernels ---------------------------------------------
// Fragment-packed B layout: index = (((h*2 + c)*NT + nt)*32 + lane)*16 + j
//   column n = nt*16 + (lane&15) ; k-row f = c*32 + ((lane&16)?16:0) + j

__global__ __launch_bounds__(256) void pack_first_kernel(
    const float* __restrict__ Wf, __bf16* __restrict__ BfP) {
  const int tid = blockIdx.x * 256 + threadIdx.x;   // 4096 = (c,nt,lane,j)
  const int j = tid & 15, lane = (tid >> 4) & 31;
  const int nt = (tid >> 9) & 3, c = (tid >> 11) & 1;
  const int f = c * 32 + ((lane & 16) ? 16 : 0) + j;
  const int h = nt * 16 + (lane & 15);
  BfP[tid] = (__bf16)Wf[h * kF + f];                // W_first[h,0,f]
}

__global__ __launch_bounds__(256) void pack_mid_kernel(
    const float* __restrict__ Wm, __bf16* __restrict__ BmP) {
  const int tid = blockIdx.x * 256 + threadIdx.x;   // 262144 = (h,c,nt,lane,j)
  const int j = tid & 15, lane = (tid >> 4) & 31;
  const int nt = (tid >> 9) & 3, c = (tid >> 11) & 1, h = tid >> 12;
  const int f = c * 32 + ((lane & 16) ? 16 : 0) + j;
  const int o = nt * 16 + (lane & 15);
  BmP[tid] = (__bf16)Wm[(size_t)o * (kH * kF) + h * kF + f];  // W_mid[o,h,f]
}

__global__ __launch_bounds__(256) void pack_last_kernel(
    const float* __restrict__ Wl, __bf16* __restrict__ BlP) {
  const int tid = blockIdx.x * 256 + threadIdx.x;   // 4M = (h,c,ntg,lane,j)
  const int j = tid & 15, lane = (tid >> 4) & 31;
  const int ntg = (tid >> 9) & 63, c = (tid >> 15) & 1, h = tid >> 16;
  const int f = c * 32 + ((lane & 16) ? 16 : 0) + j;
  const int cc = ntg * 16 + (lane & 15);
  BlP[tid] = (cc < kC) ? (__bf16)Wl[(size_t)cc * (kH * kF) + h * kF + f]
                       : (__bf16)0.0f;
}

// x[b][f][d] f32 -> xt[d][b*64+f] bf16 ; coalesced both directions via LDS.
__global__ __launch_bounds__(256) void transpose_x_kernel(
    const float* __restrict__ x, __bf16* __restrict__ xt) {
  __shared__ float tile[64][65];
  const size_t r0 = (size_t)blockIdx.x * 64;
  const int t = threadIdx.x;
#pragma unroll
  for (int i = 0; i < 16; ++i) {
    const int idx = i * 256 + t;
    const int rr = idx >> 6, d = idx & 63;
    tile[rr][d] = x[(r0 + rr) * 64 + d];
  }
  __syncthreads();
#pragma unroll
  for (int i = 0; i < 16; ++i) {
    const int idx = i * 256 + t;
    const int d = idx >> 6, rr = idx & 63;
    xt[(size_t)d * kRF + r0 + rr] = (__bf16)tile[rr][d];
  }
}

__global__ __launch_bounds__(256) void init_stats_kernel(float* __restrict__ s) {
  s[blockIdx.x * 256 + threadIdx.x] = 0.0f;         // 62*128 = 7936 exact
}

// ---------------- first layer: carry = bf16( f0 @ W_first^T ) --------------

__global__ __launch_bounds__(256) void first_layer_kernel(
    const __bf16* __restrict__ f0, const __bf16* __restrict__ BfP,
    __bf16* __restrict__ carry) {
  __shared__ __bf16 sB[4096];
  const int t = threadIdx.x, wave = t >> 5, lane = t & 31;
  const int m0 = blockIdx.x * 128;
#pragma unroll
  for (int i = 0; i < 2; ++i)
    ((uint4*)sB)[i * 256 + t] = ((const uint4*)BfP)[i * 256 + t];
  __syncthreads();

  const int mrow = m0 + wave * 16 + (lane & 15);
  v16bf a0, a1;
  load_a_frags(f0 + (size_t)mrow * kF, lane, a0, a1);

  v8f acc[4] = {};
#pragma unroll
  for (int nt = 0; nt < 4; ++nt) {
    acc[nt] = wmma_bf16(a0, load_frag_lds(sB + ((0 * 4 + nt) * 32 + lane) * 16),
                        acc[nt]);
    acc[nt] = wmma_bf16(a1, load_frag_lds(sB + ((1 * 4 + nt) * 32 + lane) * 16),
                        acc[nt]);
  }
  const int mbase = m0 + wave * 16 + ((lane & 16) ? 8 : 0);
#pragma unroll
  for (int nt = 0; nt < 4; ++nt) {
    const int n = nt * 16 + (lane & 15);
#pragma unroll
    for (int v = 0; v < 8; ++v)
      carry[(size_t)(mbase + v) * kH + n] = (__bf16)acc[nt][v];
  }
}

// -------- bilinear step: z = (row-scaled outer-product GEMM) + stats -------
// Double-buffered B slabs filled by async global->LDS DMA (ASYNCcnt), so
// next-slab copies overlap current-slab WMMA and compute ds_loads never
// queue behind copy stores.

__global__ __launch_bounds__(256) void bilinear_mid_kernel(
    const __bf16* __restrict__ carry, const __bf16* __restrict__ fslice,
    const __bf16* __restrict__ BmP, float* __restrict__ z,
    float* __restrict__ stats) {
  __shared__ __bf16 sB0[4 * 4096];     // 32 KB  (4 h-slabs, fragment-packed)
  __shared__ __bf16 sB1[4 * 4096];     // 32 KB  (double buffer)
  __shared__ __bf16 sC[128 * 72];      // carry tile, pitch 72 (bank pad)
  __shared__ float  sSum[64], sSq[64];

  const int t = threadIdx.x, wave = t >> 5, lane = t & 31;
  const int m0 = blockIdx.x * 128;
  const unsigned ldsB0 = (unsigned)(size_t)(void*)sB0;
  const unsigned ldsB1 = (unsigned)(size_t)(void*)sB1;
  const unsigned long long gB = (unsigned long long)(const void*)BmP;

  // per-thread async copy of one 32 KB slab group (8 x b128 per thread)
  auto issue_copy = [&](unsigned ldsBase, int hb) {
    unsigned long long g = gB + (size_t)hb * 32768 + (unsigned)t * 16;
    unsigned l = ldsBase + (unsigned)t * 16;
#pragma unroll
    for (int i = 0; i < 8; ++i) {
      asm volatile("global_load_async_to_lds_b128 %0, %1, off"
                   :: "v"(l), "v"(g) : "memory");
      g += 4096; l += 4096;
    }
  };

  issue_copy(ldsB0, 0);                // prologue: slab group 0 in flight

  // stage carry tile [128][64] bf16 (regular DS path)
#pragma unroll
  for (int i = 0; i < 4; ++i) {
    const int idx = i * 256 + t, row = idx >> 3, q = idx & 7;
    ((uint4*)(sC + row * 72))[q] =
        ((const uint4*)(carry + (size_t)(m0 + row) * kH))[q];
  }
  if (t < 64) { sSum[t] = 0.0f; sSq[t] = 0.0f; }

  const int rloc = wave * 16 + (lane & 15);
  v16bf a0, a1;
  load_a_frags(fslice + (size_t)(m0 + rloc) * kF, lane, a0, a1);

  v8f acc[4] = {};
  for (int hb = 0; hb < 16; ++hb) {            // 4 h per slab group
    asm volatile("s_wait_asynccnt 0x0" ::: "memory");  // my copies landed
    __syncthreads();                                    // everyone's landed
    if (hb + 1 < 16) issue_copy((hb & 1) ? ldsB0 : ldsB1, hb + 1);
    const __bf16* sB = (hb & 1) ? sB1 : sB0;

#pragma unroll
    for (int hh = 0; hh < 4; ++hh) {
      const int h = hb * 4 + hh;
      const v16bf cs =
          splat_bf16_bits(((const unsigned short*)sC)[rloc * 72 + h]);
      const v16bf a0s = a0 * cs, a1s = a1 * cs;
      const __bf16* slab = sB + hh * 4096;
#pragma unroll
      for (int nt = 0; nt < 4; ++nt) {
        acc[nt] = wmma_bf16(
            a0s, load_frag_lds(slab + ((0 * 4 + nt) * 32 + lane) * 16), acc[nt]);
        acc[nt] = wmma_bf16(
            a1s, load_frag_lds(slab + ((1 * 4 + nt) * 32 + lane) * 16), acc[nt]);
      }
    }
  }

  // write z, reduce per-column sum / sumsq (LDS atomics, then one global add)
  const int mbase = m0 + wave * 16 + ((lane & 16) ? 8 : 0);
#pragma unroll
  for (int nt = 0; nt < 4; ++nt) {
    const int n = nt * 16 + (lane & 15);
    float s = 0.0f, q = 0.0f;
#pragma unroll
    for (int v = 0; v < 8; ++v) {
      const float val = acc[nt][v];
      z[(size_t)(mbase + v) * kH + n] = val;
      s += val; q += val * val;
    }
    atomicAdd(&sSum[n], s);
    atomicAdd(&sSq[n], q);
  }
  __syncthreads();
  if (t < 64) {
    atomicAdd(&stats[t], sSum[t]);
    atomicAdd(&stats[64 + t], sSq[t]);
  }
}

// -------- BatchNorm (training mode, biased var) -> next carry (bf16) -------

__global__ __launch_bounds__(256) void bn_normalize_kernel(
    const float* __restrict__ z, const float* __restrict__ stats,
    const float* __restrict__ gamma, const float* __restrict__ beta,
    __bf16* __restrict__ carry) {
  const int tid = blockIdx.x * 256 + threadIdx.x;
  const int o = tid & 63;
  const float inv = 1.0f / (float)kB;
  const float mu  = stats[o] * inv;
  const float var = stats[64 + o] * inv - mu * mu;
  const float sc  = gamma[o] * rsqrtf(var + 1e-5f);
  const float sh  = beta[o] - mu * sc;
  carry[tid] = (__bf16)(z[tid] * sc + sh);
}

// -------- last layer: out[b][c], C=1000 (padded 1024) ----------------------

__global__ __launch_bounds__(256) void last_layer_kernel(
    const __bf16* __restrict__ carry, const __bf16* __restrict__ fslice,
    const __bf16* __restrict__ BlP, float* __restrict__ out) {
  __shared__ __bf16 sB[4 * 4096];      // 4 h-slabs for this 64-col slice
  __shared__ __bf16 sC[128 * 72];

  const int t = threadIdx.x, wave = t >> 5, lane = t & 31;
  const int m0 = blockIdx.x * 128;
  const int gy = blockIdx.y;           // class-column slice (64 cols)

#pragma unroll
  for (int i = 0; i < 4; ++i) {
    const int idx = i * 256 + t, row = idx >> 3, q = idx & 7;
    ((uint4*)(sC + row * 72))[q] =
        ((const uint4*)(carry + (size_t)(m0 + row) * kH))[q];
  }

  const int rloc = wave * 16 + (lane & 15);
  v16bf a0, a1;
  load_a_frags(fslice + (size_t)(m0 + rloc) * kF, lane, a0, a1);

  v8f acc[4] = {};
  for (int hb = 0; hb < 16; ++hb) {
    __syncthreads();
    // 8 contiguous runs of 2048 elems: run = (hh*2 + c)
#pragma unroll
    for (int i = 0; i < 4; ++i) {
      const int idx = i * 256 + t;       // 0..1023 uint4
      const int run = idx >> 7, q = idx & 127;
      const int h = hb * 4 + (run >> 1), c = run & 1;
      const uint4* src =
          (const uint4*)(BlP + ((size_t)((h * 2 + c) * 64 + 4 * gy)) * 512);
      ((uint4*)(sB + run * 2048))[q] = src[q];
    }
    // prefetch next slab group into L2/L0 while we compute
    if (hb + 1 < 16) {
      const int hn = (hb + 1) * 4;
      __builtin_prefetch(BlP + ((size_t)(hn * 2) * 64 + 4 * gy) * 512 + t * 8,
                         0, 1);
    }
    __syncthreads();

#pragma unroll
    for (int hh = 0; hh < 4; ++hh) {
      const int h = hb * 4 + hh;
      const v16bf cs =
          splat_bf16_bits(((const unsigned short*)sC)[rloc * 72 + h]);
      const v16bf a0s = a0 * cs, a1s = a1 * cs;
      const __bf16* slab = sB + hh * 4096;
#pragma unroll
      for (int nt = 0; nt < 4; ++nt) {
        acc[nt] = wmma_bf16(
            a0s, load_frag_lds(slab + 0 * 2048 + (nt * 32 + lane) * 16),
            acc[nt]);
        acc[nt] = wmma_bf16(
            a1s, load_frag_lds(slab + 1 * 2048 + (nt * 32 + lane) * 16),
            acc[nt]);
      }
    }
  }

  const int mbase = m0 + wave * 16 + ((lane & 16) ? 8 : 0);
#pragma unroll
  for (int nt = 0; nt < 4; ++nt) {
    const int c = gy * 64 + nt * 16 + (lane & 15);
    if (c < kC) {
#pragma unroll
      for (int v = 0; v < 8; ++v)
        out[(size_t)(mbase + v) * kC + c] = acc[nt][v];
    }
  }
}

// ---------------- launch ---------------------------------------------------

extern "C" void kernel_launch(void* const* d_in, const int* in_sizes, int n_in,
                              void* d_out, int out_size, void* d_ws,
                              size_t ws_size, hipStream_t stream) {
  (void)in_sizes; (void)n_in; (void)out_size; (void)ws_size;
  const float* x  = (const float*)d_in[0];
  const float* Wf = (const float*)d_in[1];
  const float* Wm = (const float*)d_in[2];
  const float* Wl = (const float*)d_in[3];
  const float* gamma = (const float*)d_in[4];
  const float* beta  = (const float*)d_in[5];
  float* out = (float*)d_out;

  char* ws = (char*)d_ws;  // assumes ws_size >= ~150 MB
  __bf16* xt    = (__bf16*)(ws + XT_OFF);
  __bf16* BmP   = (__bf16*)(ws + BM_OFF);
  __bf16* BlP   = (__bf16*)(ws + BL_OFF);
  __bf16* BfP   = (__bf16*)(ws + BF_OFF);
  float*  z     = (float*)(ws + Z_OFF);
  __bf16* carry = (__bf16*)(ws + CR_OFF);
  float*  stats = (float*)(ws + ST_OFF);

  pack_first_kernel<<<16, 256, 0, stream>>>(Wf, BfP);
  pack_mid_kernel<<<1024, 256, 0, stream>>>(Wm, BmP);
  pack_last_kernel<<<16384, 256, 0, stream>>>(Wl, BlP);
  transpose_x_kernel<<<16384, 256, 0, stream>>>(x, xt);
  init_stats_kernel<<<31, 256, 0, stream>>>(stats);

  first_layer_kernel<<<kB / 128, 256, 0, stream>>>(xt /*slice 0*/, BfP, carry);

  for (int s = 0; s < kSteps; ++s) {
    bilinear_mid_kernel<<<kB / 128, 256, 0, stream>>>(
        carry, xt + (size_t)s * kRF, BmP, z, stats + (size_t)s * 128);
    bn_normalize_kernel<<<(kB * kH) / 256, 256, 0, stream>>>(
        z, stats + (size_t)s * 128, gamma, beta, carry);
  }

  last_layer_kernel<<<dim3(kB / 128, kCp / 64), 256, 0, stream>>>(
      carry, xt + (size_t)(kD - 1) * kRF, BlP, out);
}
